// Model_79456894976282
// MI455X (gfx1250) — matrix operand order repaired and checked by reading, not compiled, not run on hardware
//
#include <hip/hip_runtime.h>
#include <hip/hip_bf16.h>

typedef _Float16 half_t;
typedef _Float16 v16h __attribute__((ext_vector_type(16)));
typedef _Float16 v8h  __attribute__((ext_vector_type(8)));
typedef float    v8f  __attribute__((ext_vector_type(8)));

#define BTOT   8192
#define TSTEPS 30
#define NWARM  24
#define HDIM   128
#define FINC   8
#define KCAT   160   // 8 input + 128 hidden, padded to 5 K-tiles of 32
#define NZG    512   // 4*H gate width
#define MLPW   128
#define BM     32    // batch rows per workgroup
#define NTHR   512   // 16 waves of 32 -> 4 waves/SIMD32

__device__ __forceinline__ float sigm(float x) { return 1.0f / (1.0f + __expf(-x)); }
__device__ __forceinline__ float tanh_fast(float x) {
    x = fminf(fmaxf(x, -15.0f), 15.0f);
    float e = __expf(2.0f * x);
    return (e - 1.0f) / (e + 1.0f);
}

__device__ __forceinline__ v8f wmma_f16(v16h a, v16h b, v8f c) {
    // D = A(16x32 f16) * B(32x16 f16) + C(16x16 f32)
    return __builtin_amdgcn_wmma_f32_16x16x32_f16(
        /*neg_a=*/false, a, /*neg_b=*/false, b,
        /*c_mod=*/(short)0, c, /*reuse_a=*/false, /*reuse_b=*/false);
}

// A-fragment (contiguous k order, same as loadB): lane holds K = [k0, k0+16)
// where k0 = ktile*32 + (lane<16 ? 0 : 16). Two 16-byte LDS chunks.
__device__ __forceinline__ v16h loadA(const half_t* rowBase, int k0) {
    union { v16h v; v8h h[2]; } u;
    u.h[0] = *(const v8h*)(rowBase + k0);
    u.h[1] = *(const v8h*)(rowBase + k0 + 8);
    return u.v;
}

// B-fragment (contiguous k order, same as loadA): lane = column n (lane%16),
// K = ktile*32 + (lane<16?0:16), 16 contiguous halfs (32 bytes).
__device__ __forceinline__ v16h loadB(const half_t* rowBase, int k0) {
    return *(const v16h*)(rowBase + k0);
}

__global__ __launch_bounds__(NTHR, 1)
void lstm_forecast_kernel(
    const float* __restrict__ f0, const float* __restrict__ f1,
    const float* __restrict__ f2, const float* __restrict__ f3,
    const float* __restrict__ f4, const float* __restrict__ f5,
    const float* __restrict__ f6,
    const float* __restrict__ irr,
    const float* __restrict__ Wi,  const float* __restrict__ Wh,
    const float* __restrict__ b,
    const float* __restrict__ W1,  const float* __restrict__ b1,
    const float* __restrict__ W2,  const float* __restrict__ b2,
    const float* __restrict__ Wout, const float* __restrict__ bout,
    float* __restrict__ out)
{
    // ---- LDS: ~254 KB of the 320 KB/WGP budget ----
    __shared__ __align__(64) half_t sWz[NZG][KCAT];   // 160 KB  [Wi;Wh;0]^T, [n][k]
    __shared__ __align__(64) half_t sW1[MLPW][HDIM];  //  32 KB  W1^T
    __shared__ __align__(64) half_t sW2[MLPW][MLPW];  //  32 KB  W2^T
    __shared__ __align__(64) half_t sA [BM][KCAT];    //  10 KB  [x(8)|h(128)|pad(24)]
    __shared__ __align__(64) half_t sO1[BM][MLPW];    //   8 KB
    __shared__ __align__(64) half_t sO2[BM][MLPW];    //   8 KB
    __shared__ float sB[NZG];                         //   2 KB
    __shared__ float sB1[MLPW];
    __shared__ float sB2[MLPW];
    __shared__ float sWo[MLPW];
    __shared__ float sPred[BM];

    const int tid  = threadIdx.x;
    const int lane = tid & 31;
    const int wv   = tid >> 5;       // wave 0..15
    const int h8   = wv & 7;         // hidden-unit tile owned (all 4 gates)
    const int mt   = wv >> 3;        // M-tile owned (0 or 1)
    const int lh   = lane & 15;      // N/M sub-index
    const int hi   = lane >> 4;      // half-wave select
    const int row0 = blockIdx.x * BM;

    // ---- stage weights into LDS (f32 -> f16, transposed) ----
    // Traverse k-major/n-fastest so GLOBAL reads are fully coalesced
    // (Wi/Wh/W1/W2 read linearly); the transpose scatter lands on cheap LDS stores.
    for (int idx = tid; idx < (FINC + HDIM) * NZG; idx += NTHR) {
        int k = idx / NZG, n = idx - k * NZG;             // contiguous global read
        float v = (k < FINC) ? Wi[idx] : Wh[idx - FINC * NZG];
        sWz[n][k] = (half_t)v;
    }
    for (int idx = tid; idx < NZG * (KCAT - FINC - HDIM); idx += NTHR) {
        int n = idx / (KCAT - FINC - HDIM);
        int k = (FINC + HDIM) + idx - n * (KCAT - FINC - HDIM);
        sWz[n][k] = (half_t)0.0f;                         // zero pad: keeps WMMA NaN-free
    }
    for (int idx = tid; idx < HDIM * MLPW; idx += NTHR) {
        int k = idx / MLPW, n = idx - k * MLPW;           // contiguous global read
        sW1[n][k] = (half_t)W1[idx];
        sW2[n][k] = (half_t)W2[idx];
    }
    for (int idx = tid; idx < NZG; idx += NTHR) sB[idx] = b[idx];
    for (int idx = tid; idx < MLPW; idx += NTHR) {
        sB1[idx] = b1[idx];
        sB2[idx] = b2[idx];
        sWo[idx] = Wout[idx];
    }
    for (int idx = tid; idx < BM * KCAT; idx += NTHR)
        (&sA[0][0])[idx] = (half_t)0.0f;   // h0 = 0, pad = 0
    const float boutv = bout[0];

    // per-thread input-fill assignment (BM*FINC == 256 threads participate)
    const int m_in = (tid & 255) >> 3;
    const int k_in = tid & 7;
    const float* inBase;
    switch (k_in) {
        case 0: inBase = f0; break; case 1: inBase = f1; break;
        case 2: inBase = f2; break; case 3: inBase = f3; break;
        case 4: inBase = f4; break; case 5: inBase = f5; break;
        case 6: inBase = f6; break; default: inBase = irr; break;
    }
    const float* featPtr = (k_in < 7)
        ? inBase + (size_t)(row0 + m_in) * TSTEPS
        : inBase + (size_t)(row0 + m_in) * NWARM;

    // pred-stage assignment: 8 lanes per row, wave-local reduction groups
    const int m_pr = tid >> 3;       // row (valid when tid < BM*8)
    const int p_pr = tid & 7;        // 16-element slice within the row

    // gate bias values for this wave's columns (hidden units j = 16*h8 + lh)
    const float bI = sB[          h8 * 16 + lh];
    const float bF = sB[1 * HDIM + h8 * 16 + lh];
    const float bG = sB[2 * HDIM + h8 * 16 + lh];
    const float bO = sB[3 * HDIM + h8 * 16 + lh];
    const float b1v = sB1[h8 * 16 + lh];
    const float b2v = sB2[h8 * 16 + lh];

    v8f cfrag = {};   // persistent cell state, C-fragment layout (this wave's M-tile)
    __syncthreads();

    for (int t = 0; t < TSTEPS; ++t) {
        // ---- fill input columns 0..7 of sA ----
        if (tid < BM * FINC) {
            float v;
            if (k_in < 7)            v = featPtr[t];
            else if (t < NWARM)      v = featPtr[t];
            else                     v = sPred[m_in];        // autoregressive feedback
            sA[m_in][k_in] = (half_t)v;
        }
        __syncthreads();  // A: inputs + h ready

        // ---- z = [x;h] @ [Wi;Wh] + b ----
        // wave wv: gates i/f/g/o of hidden tile h8, M-tile mt
        v8f az[4];
        az[0] = v8f{bI,bI,bI,bI,bI,bI,bI,bI};
        az[1] = v8f{bF,bF,bF,bF,bF,bF,bF,bF};
        az[2] = v8f{bG,bG,bG,bG,bG,bG,bG,bG};
        az[3] = v8f{bO,bO,bO,bO,bO,bO,bO,bO};
        #pragma unroll
        for (int kt = 0; kt < 5; ++kt) {
            const int ka = kt * 32 + hi * 16;
            const int kb = kt * 32 + hi * 16;
            v16h a = loadA(&sA[mt * 16 + lh][0], ka);
            #pragma unroll
            for (int g = 0; g < 4; ++g) {
                const int n = g * HDIM + h8 * 16 + lh;
                v16h bf = loadB(&sWz[n][0], kb);
                az[g] = wmma_f16(a, bf, az[g]);
            }
        }
        __syncthreads();  // B: all reads of sA complete

        // ---- gate math (wave-local) + write new h into sA cols 8..135 ----
        #pragma unroll
        for (int r = 0; r < 8; ++r) {
            float iv = sigm(az[0][r]);
            float fv = sigm(az[1][r]);
            float gv = tanh_fast(az[2][r]);
            float ov = sigm(az[3][r]);
            float c2 = fv * cfrag[r] + iv * gv;
            cfrag[r] = c2;
            float h2 = ov * tanh_fast(c2);
            sA[mt * 16 + hi * 8 + r][FINC + h8 * 16 + lh] = (half_t)h2;
        }
        __syncthreads();  // C: new h visible

        // ---- MLP layer 1: relu(h @ W1 + b1); wave: N-tile h8, M-tile mt ----
        v8f o1 = v8f{b1v,b1v,b1v,b1v,b1v,b1v,b1v,b1v};
        #pragma unroll
        for (int kt = 0; kt < 4; ++kt) {
            v16h a  = loadA(&sA[mt * 16 + lh][FINC], kt * 32 + hi * 16);
            v16h bf = loadB(&sW1[h8 * 16 + lh][0],   kt * 32 + hi * 16);
            o1 = wmma_f16(a, bf, o1);
        }
        #pragma unroll
        for (int r = 0; r < 8; ++r)
            sO1[mt * 16 + hi * 8 + r][h8 * 16 + lh] = (half_t)fmaxf(o1[r], 0.0f);
        __syncthreads();  // D

        // ---- MLP layer 2: relu(o1 @ W2 + b2) ----
        v8f o2 = v8f{b2v,b2v,b2v,b2v,b2v,b2v,b2v,b2v};
        #pragma unroll
        for (int kt = 0; kt < 4; ++kt) {
            v16h a  = loadA(&sO1[mt * 16 + lh][0], kt * 32 + hi * 16);
            v16h bf = loadB(&sW2[h8 * 16 + lh][0], kt * 32 + hi * 16);
            o2 = wmma_f16(a, bf, o2);
        }
        #pragma unroll
        for (int r = 0; r < 8; ++r)
            sO2[mt * 16 + hi * 8 + r][h8 * 16 + lh] = (half_t)fmaxf(o2[r], 0.0f);
        __syncthreads();  // E

        // ---- pred = o2 @ Wout + bout ----
        // 8 lanes per row; each lane: 16 products from contiguous LDS chunks,
        // then wave-local xor-shuffle reduction (parts of a row = adjacent lanes).
        if (tid < BM * 8) {
            const half_t* rp = &sO2[m_pr][p_pr * 16];
            const float*  wp = &sWo[p_pr * 16];
            float s = 0.0f;
            #pragma unroll
            for (int k = 0; k < 16; ++k)
                s += (float)rp[k] * wp[k];
            s += __shfl_xor(s, 1, 32);
            s += __shfl_xor(s, 2, 32);
            s += __shfl_xor(s, 4, 32);
            if (p_pr == 0) {
                float pv = s + boutv;
                out[(size_t)(row0 + m_pr) * TSTEPS + t] = pv;
                sPred[m_pr] = pv;
            }
        }
        __syncthreads();  // F: pred visible for next step's input fill
    }
}

extern "C" void kernel_launch(void* const* d_in, const int* in_sizes, int n_in,
                              void* d_out, int out_size, void* d_ws, size_t ws_size,
                              hipStream_t stream) {
    // setup_inputs order: [0..7]=WX (unused by reference), [8..14]=TL features,
    // 15=irradiance_in, 16=Wi, 17=Wh, 18=b, 19=W1, 20=b1, 21=W2, 22=b2, 23=Wout, 24=bout
    const float* hour_sin    = (const float*)d_in[8];
    const float* hour_cos    = (const float*)d_in[9];
    const float* year_sin    = (const float*)d_in[10];
    const float* year_cos    = (const float*)d_in[11];
    const float* elev_angle  = (const float*)d_in[12];
    const float* azimuth_sin = (const float*)d_in[13];
    const float* azimuth_cos = (const float*)d_in[14];
    const float* irr  = (const float*)d_in[15];
    const float* Wi   = (const float*)d_in[16];
    const float* Wh   = (const float*)d_in[17];
    const float* b    = (const float*)d_in[18];
    const float* W1   = (const float*)d_in[19];
    const float* b1   = (const float*)d_in[20];
    const float* W2   = (const float*)d_in[21];
    const float* b2   = (const float*)d_in[22];
    const float* Wout = (const float*)d_in[23];
    const float* bout = (const float*)d_in[24];
    float* out = (float*)d_out;

    hipLaunchKernelGGL(lstm_forecast_kernel,
                       dim3(BTOT / BM), dim3(NTHR), 0, stream,
                       hour_sin, hour_cos, year_sin, year_cos,
                       elev_angle, azimuth_sin, azimuth_cos,
                       irr, Wi, Wh, b, W1, b1, W2, b2, Wout, bout, out);
}
